// FullHierEnergy_49512382988663
// MI455X (gfx1250) — compile-verified, matrix operand order checked by reading, not run on hardware
//
#include <hip/hip_runtime.h>
#include <hip/hip_bf16.h>

// ---------------- problem constants ----------------
#define BB     4096          // batch
#define TSENT  80
#define NSENT  (BB*TSENT)    // flattened sentence rows (pad row at NSENT)
#define CTXW   10            // LSTM seq len per side
#define NPOS   20            // attention positions
#define DD     300           // word dim
#define WES    304           // padded we_bf row stride (8-elem aligned)
#define XPAD   304           // K offset where h-part starts in fused [x|h] axis
#define RR     200           // LSTM hidden
#define H4     800           // 4*R
#define KPAD   512           // padded K for [x|h] GEMM (304 + 200 + 8)
#define HAA    100           // attention hidden
#define TDD    256           // type dim
#define NTY    113
#define NTOK   250
#define PLL    3
#define K_ATT  400           // 2R
#define CTXS   416           // padded ctx row stride (13 K-tiles)
#define K_DENSE 700
#define KD_PAD 704
#define VOC    100000

typedef __attribute__((ext_vector_type(16))) __bf16        v16bf;
typedef __attribute__((ext_vector_type(8)))  float         v8f;
typedef __attribute__((ext_vector_type(8)))  unsigned int  u8v;

__device__ __forceinline__ unsigned short f2bf(float f) {
    unsigned u = __builtin_bit_cast(unsigned, f);
    u += 0x7FFFu + ((u >> 16) & 1u);
    return (unsigned short)(u >> 16);
}
__device__ __forceinline__ float bf2f(unsigned short h) {
    return __builtin_bit_cast(float, ((unsigned)h) << 16);
}
__device__ __forceinline__ unsigned pack2(float a, float b) {
    return (unsigned)f2bf(a) | ((unsigned)f2bf(b) << 16);
}

// ---- WMMA fragment assembly from LDS (ISA 7.12.2 16-bit layouts) ----
// Both operands staged K-contiguous per row/column -> lowers to ds_load_2addr_b64 pairs.
__device__ __forceinline__ v16bf frag_K(const unsigned short* lds, int ldw, int lane) {
    int half = lane >> 4, m = lane & 15;
    u8v p;
#pragma unroll
    for (int v = 0; v < 8; ++v) {
        int k = ((v < 4) ? 0 : 16) + half * 8 + 2 * (v & 3);
        unsigned lo = lds[m * ldw + k];
        unsigned hi = lds[m * ldw + k + 1];
        p[v] = lo | (hi << 16);
    }
    return __builtin_bit_cast(v16bf, p);
}
#define WMMA_BF16(A, Bf, C) \
    __builtin_amdgcn_wmma_f32_16x16x32_bf16(false, (A), false, (Bf), (short)0, (C), false, false)

// ---------------- init / conversion kernels ----------------
__global__ void zero_u32(unsigned* p, int n) {
    int i = blockIdx.x * blockDim.x + threadIdx.x;
    if (i < n) p[i] = 0u;
}
// word_embed -> bf16 with padded row stride 304 (cols 300..303 zero)
__global__ void conv_embed(const float* __restrict__ src, unsigned* __restrict__ dst) {
    int i = blockIdx.x * blockDim.x + threadIdx.x;         // pairs: VOC*152
    if (i >= VOC * (WES / 2)) return;
    int row = i / (WES / 2), kp = i - row * (WES / 2);
    int K = 2 * kp;
    float a = 0.f, b = 0.f;
    if (K < DD) { a = src[row * DD + K]; b = src[row * DD + K + 1]; }
    dst[i] = pack2(a, b);
}
// WcombT[dir][800][512]: column n holds K-contig [Wx(:,n)(300); 0(4); Wh(:,n)(200); 0(8)] bf16
__global__ void conv_weightsT(const float* __restrict__ Wx_f, const float* __restrict__ Wh_f,
                              const float* __restrict__ Wx_b, const float* __restrict__ Wh_b,
                              unsigned* __restrict__ WcombT) {
    int i = blockIdx.x * blockDim.x + threadIdx.x;         // pairs: 2*800*256
    if (i >= 2 * H4 * (KPAD / 2)) return;
    int d = i / (H4 * (KPAD / 2));
    int rem = i - d * (H4 * (KPAD / 2));
    int n = rem / (KPAD / 2), kp = rem - n * (KPAD / 2);
    int K = 2 * kp;
    const float* Wx = d ? Wx_b : Wx_f;
    const float* Wh = d ? Wh_b : Wh_f;
    float a = 0.f, b = 0.f;
    if (K < DD) {
        a = Wx[K * H4 + n];
        if (K + 1 < DD) b = Wx[(K + 1) * H4 + n];
    } else if (K >= XPAD && K < XPAD + RR) {
        a = Wh[(K - XPAD) * H4 + n];
        b = Wh[(K + 1 - XPAD) * H4 + n];
    }
    WcombT[i] = pack2(a, b);
}
// dWbfT[256][704]: column n holds K-contiguous dense_W(:,n), rows>=700 zero
__global__ void conv_denseT(const float* __restrict__ W, unsigned* __restrict__ dst) {
    int i = blockIdx.x * blockDim.x + threadIdx.x;         // pairs: 256*352
    if (i >= TDD * (KD_PAD / 2)) return;
    int n = i / (KD_PAD / 2), kp = i - n * (KD_PAD / 2);
    int K = 2 * kp;
    float a = 0.f, b = 0.f;
    if (K < K_DENSE) { a = W[K * TDD + n]; b = W[(K + 1) * TDD + n]; }
    dst[i] = pack2(a, b);
}
// h1bfT[128][416]: column n holds K-contiguous h1(:,n), zero-padded (cols>=100, K>=400)
__global__ void conv_h1T(const float* __restrict__ h1, unsigned* __restrict__ dst) {
    int i = blockIdx.x * blockDim.x + threadIdx.x;         // pairs: 128*208
    if (i >= 128 * (CTXS / 2)) return;
    int n = i / (CTXS / 2), kp = i - n * (CTXS / 2);
    int K = 2 * kp;
    float a = 0.f, b = 0.f;
    if (n < HAA && K < K_ATT) { a = h1[K * HAA + n]; b = h1[(K + 1) * HAA + n]; }
    dst[i] = pack2(a, b);
}

// ---------------- small feature kernels ----------------
__global__ void f1_kernel(const int* __restrict__ input_data, const int* __restrict__ mentIdx,
                          const float* __restrict__ mentLen, const float* __restrict__ word_embed,
                          float* __restrict__ f1, unsigned short* __restrict__ feat_bf) {
    int idx = blockIdx.x * blockDim.x + threadIdx.x;
    if (idx >= BB * DD) return;
    int b = idx / DD, d = idx - b * DD;
    float s = 0.f;
#pragma unroll
    for (int j = 0; j < 5; ++j) {
        int sent = mentIdx[b * 5 + j];
        if ((unsigned)sent < (unsigned)NSENT) {
            int tk = input_data[sent];
            s += word_embed[tk * DD + d];
        }
    }
    float val = s / mentLen[b];
    f1[idx] = val;
    feat_bf[(size_t)b * KD_PAD + d] = f2bf(val);
}
__global__ void fhm_kernel(const float* __restrict__ f1, const float* __restrict__ h_m,
                           float* __restrict__ fhm) {
    int idx = blockIdx.x * blockDim.x + threadIdx.x;
    if (idx >= BB * HAA) return;
    int b = idx / HAA, h = idx - b * HAA;
    float s = 0.f;
    for (int k = 0; k < DD; ++k) s += f1[b * DD + k] * h_m[k * HAA + h];
    fhm[idx] = s;
}

// ---------------- LSTM step GEMM: gates = gather(x)|h_bf @ [Wx;Wh] (bf16 WMMA) ----------------
// grid: (B/64, 800/160, 4 variants), block 128 (4 waves); wave = 16 rows x 160 cols (10 tiles)
__global__ void lstm_gates_kernel(int t,
                                  const int* __restrict__ input_data,
                                  const int* __restrict__ idxL, const int* __restrict__ idxR,
                                  const float* __restrict__ lenL, const float* __restrict__ lenR,
                                  const unsigned short* __restrict__ we_bf,
                                  const unsigned short* __restrict__ WcombT,
                                  const unsigned short* __restrict__ h_bf,
                                  float* __restrict__ gates) {
    const int tid = threadIdx.x, lane = tid & 31, w = tid >> 5;
    const int v = blockIdx.z, dir = v & 1, side = v >> 1;
    const int m_base = blockIdx.x * 64;
    const int n0 = blockIdx.y * 160;
    const int* idxArr = side ? idxR : idxL;
    const float* lenArr = side ? lenR : lenL;
    const unsigned short* WdT = WcombT + (size_t)dir * H4 * KPAD;

    __shared__ unsigned short As[64][40];
    __shared__ unsigned short BsT[160][40];
    __shared__ int tok[64];

    if (tid < 64) {
        int b = m_base + tid;
        int len = (int)lenArr[b];
        len = len < 0 ? 0 : (len > CTXW ? CTXW : len);
        int te = dir ? ((t < len) ? (len - 1 - t) : t) : t;
        int sent = idxArr[b * CTXW + te];
        tok[tid] = ((unsigned)sent < (unsigned)NSENT) ? input_data[sent] : -1;
    }
    __syncthreads();

    v8f acc[10] = {};
    for (int kt = 0; kt < KPAD / 32; ++kt) {
        int k0 = kt * 32;
        // A: 64 rows x 4 octs (uint4 = 8 bf16) -> 256 octs, 2 per thread
#pragma unroll
        for (int i = 0; i < 2; ++i) {
            int idx = tid + i * 128;
            int r = idx >> 2, ko = idx & 3;
            int K = k0 + 8 * ko;
            uint4 val = make_uint4(0u, 0u, 0u, 0u);
            if (K < XPAD) {
                int tk = tok[r];
                if (tk >= 0) val = *(const uint4*)(we_bf + (size_t)tk * WES + K);
            } else if (K < XPAD + RR) {
                val = *(const uint4*)(h_bf + ((size_t)v * BB + (m_base + r)) * RR + (K - XPAD));
            }
            *(uint4*)&As[r][8 * ko] = val;
        }
        // B: 160 cols x 4 octs -> 640 octs, 5 per thread (branch-free, K-contiguous)
#pragma unroll
        for (int i = 0; i < 5; ++i) {
            int idx = tid + i * 128;
            int n = idx >> 2, ko = idx & 3;
            *(uint4*)&BsT[n][8 * ko] =
                *(const uint4*)(WdT + (size_t)(n0 + n) * KPAD + k0 + 8 * ko);
        }
        __syncthreads();
        v16bf af = frag_K(&As[w * 16][0], 40, lane);
#pragma unroll
        for (int j = 0; j < 10; ++j) {
            v16bf bf = frag_K(&BsT[16 * j][0], 40, lane);
            acc[j] = WMMA_BF16(af, bf, acc[j]);
        }
        __syncthreads();
    }
    int half = lane >> 4, nn = lane & 15;
    float* gout = gates + (size_t)v * BB * H4;
#pragma unroll
    for (int j = 0; j < 10; ++j) {
#pragma unroll
        for (int e = 0; e < 8; ++e) {
            int grow = m_base + w * 16 + e + 8 * half;
            gout[(size_t)grow * H4 + n0 + 16 * j + nn] = acc[j][e];
        }
    }
}

// pointwise LSTM update: f32 master state + bf16 mirror + ctx (padded stride) write
__global__ void lstm_point_kernel(int t,
                                  const float* __restrict__ lenL, const float* __restrict__ lenR,
                                  const float* __restrict__ b_f, const float* __restrict__ b_b,
                                  const float* __restrict__ gates,
                                  float* __restrict__ hstate, float* __restrict__ cstate,
                                  unsigned short* __restrict__ h_bf,
                                  unsigned short* __restrict__ ctx) {
    int idx = blockIdx.x * blockDim.x + threadIdx.x;
    if (idx >= 4 * BB * RR) return;
    int v = idx / (BB * RR);
    int rem = idx - v * (BB * RR);
    int b = rem / RR;
    int j = rem - b * RR;
    int dir = v & 1, side = v >> 1;
    const float* lenArr = side ? lenR : lenL;
    int len = (int)lenArr[b];
    len = len < 0 ? 0 : (len > CTXW ? CTXW : len);
    const float* bias = dir ? b_b : b_f;
    const float* g = gates + ((size_t)v * BB + b) * H4;
    float gi = g[j] + bias[j];
    float gf = g[RR + j] + bias[RR + j];
    float gg = g[2 * RR + j] + bias[2 * RR + j];
    float go = g[3 * RR + j] + bias[3 * RR + j];
    float si = 1.f / (1.f + expf(-gi));
    float sf = 1.f / (1.f + expf(-gf));
    float so = 1.f / (1.f + expf(-go));
    float tg = tanhf(gg);
    int sidx = (v * BB + b) * RR + j;
    float c_new = sf * cstate[sidx] + si * tg;
    float h_new = so * tanhf(c_new);
    bool valid = (t < len);
    float h_keep = hstate[sidx];
    if (valid) { cstate[sidx] = c_new; hstate[sidx] = h_new; h_keep = h_new; }
    h_bf[sidx] = f2bf(h_keep);
    float outv = valid ? h_new : 0.f;
    int t_out = dir ? ((t < len) ? (len - 1 - t) : t) : t;
    int p = side * CTXW + t_out;
    ctx[((size_t)b * NPOS + p) * CTXS + dir * RR + j] = f2bf(outv);
}

// ---------------- attention scores: relu(ctx@h1 + fhm) . h2 ----------------
__global__ void attn_score_kernel(const unsigned short* __restrict__ ctx,
                                  const unsigned short* __restrict__ h1bfT,
                                  const float* __restrict__ h2,
                                  const float* __restrict__ fhm, float* __restrict__ scores) {
    const int tid = threadIdx.x, lane = tid & 31, w = tid >> 5;
    const int m0 = blockIdx.x * 16;
    __shared__ unsigned short As[16][40];
    __shared__ unsigned short BsT[128][40];
    __shared__ float ssc[16];
    if (tid < 16) ssc[tid] = 0.f;

    v8f acc0 = {}, acc1 = {};
    for (int kt = 0; kt < CTXS / 32; ++kt) {   // 13, branch-free (padded buffers)
        int k0 = kt * 32;
        if (tid < 64) {   // A: 16 rows x 4 octs = 64
            int r = tid >> 2, ko = tid & 3;
            *(uint4*)&As[r][8 * ko] =
                *(const uint4*)(ctx + (size_t)(m0 + r) * CTXS + k0 + 8 * ko);
        }
#pragma unroll
        for (int i = 0; i < 4; ++i) {  // B: 128 cols x 4 octs = 512, 4 per thread
            int idx = tid + i * 128;
            int n = idx >> 2, ko = idx & 3;
            *(uint4*)&BsT[n][8 * ko] =
                *(const uint4*)(h1bfT + (size_t)n * CTXS + k0 + 8 * ko);
        }
        __syncthreads();
        v16bf af = frag_K(&As[0][0], 40, lane);
        v16bf b0 = frag_K(&BsT[w * 16][0], 40, lane);
        acc0 = WMMA_BF16(af, b0, acc0);
        v16bf b1 = frag_K(&BsT[(4 + w) * 16][0], 40, lane);
        acc1 = WMMA_BF16(af, b1, acc1);
        __syncthreads();
    }
    int half = lane >> 4, nn = lane & 15;
#pragma unroll
    for (int e = 0; e < 8; ++e) {
        int row = e + 8 * half;
        int gr = m0 + row;
        int b = gr / NPOS;
        int c0 = w * 16 + nn;
        if (c0 < HAA) {
            float val = acc0[e] + fhm[b * HAA + c0];
            val = val > 0.f ? val : 0.f;
            atomicAdd(&ssc[row], val * h2[c0]);
        }
        int c1 = 64 + w * 16 + nn;
        if (c1 < HAA) {
            float val = acc1[e] + fhm[b * HAA + c1];
            val = val > 0.f ? val : 0.f;
            atomicAdd(&ssc[row], val * h2[c1]);
        }
    }
    __syncthreads();
    if (tid < 16) scores[m0 + tid] = ssc[tid];
}

__global__ void softmax20_kernel(const float* __restrict__ scores, float* __restrict__ attw) {
    int b = blockIdx.x;
    __shared__ float s[NPOS];
    int tid = threadIdx.x;
    if (tid < NPOS) s[tid] = scores[b * NPOS + tid];
    __syncthreads();
    if (tid == 0) {
        float mx = s[0];
        for (int i = 1; i < NPOS; ++i) mx = fmaxf(mx, s[i]);
        float sum = 0.f;
        for (int i = 0; i < NPOS; ++i) { float e = expf(s[i] - mx); s[i] = e; sum += e; }
        float inv = 1.f / sum;
        for (int i = 0; i < NPOS; ++i) attw[b * NPOS + i] = s[i] * inv;
    }
}

// ctx_feat -> cols 300..699 of padded bf16 feature row
__global__ void ctxfeat_kernel(const unsigned short* __restrict__ ctx,
                               const float* __restrict__ attw,
                               unsigned short* __restrict__ feat_bf) {
    int idx = blockIdx.x * blockDim.x + threadIdx.x;
    if (idx >= BB * 2 * RR) return;
    int b = idx / (2 * RR), r = idx - b * (2 * RR);
    float s = 0.f;
#pragma unroll
    for (int t = 0; t < NPOS; ++t)
        s += attw[b * NPOS + t] * bf2f(ctx[((size_t)b * NPOS + t) * CTXS + r]);
    feat_bf[(size_t)b * KD_PAD + DD + r] = f2bf(s);
}

// ---------------- dense: ent_bf = l2norm(relu(feat@W + b)) ----------------
__global__ void dense_kernel(const unsigned short* __restrict__ feat_bf,
                             const unsigned short* __restrict__ dWbfT,
                             const float* __restrict__ dense_b,
                             unsigned short* __restrict__ ent_bf) {
    const int tid = threadIdx.x, lane = tid & 31, w = tid >> 5;
    const int m0 = blockIdx.x * 16;
    __shared__ unsigned short As[16][40];
    __shared__ unsigned short BsT[256][40];
    __shared__ float ssq[16];
    if (tid < 16) ssq[tid] = 0.f;
    v8f acc[4] = {};
    for (int kt = 0; kt < KD_PAD / 32; ++kt) {  // 22, branch-free
        int k0 = kt * 32;
        if (tid < 64) {   // A: 16 rows x 4 octs = 64
            int r = tid >> 2, ko = tid & 3;
            *(uint4*)&As[r][8 * ko] =
                *(const uint4*)(feat_bf + (size_t)(m0 + r) * KD_PAD + k0 + 8 * ko);
        }
#pragma unroll
        for (int i = 0; i < 8; ++i) {  // B: 256 cols x 4 octs = 1024, 8 per thread
            int idx = tid + i * 128;
            int n = idx >> 2, ko = idx & 3;
            *(uint4*)&BsT[n][8 * ko] =
                *(const uint4*)(dWbfT + (size_t)n * KD_PAD + k0 + 8 * ko);
        }
        __syncthreads();
        v16bf af = frag_K(&As[0][0], 40, lane);
#pragma unroll
        for (int j = 0; j < 4; ++j) {
            v16bf bf = frag_K(&BsT[(w + 4 * j) * 16][0], 40, lane);
            acc[j] = WMMA_BF16(af, bf, acc[j]);
        }
        __syncthreads();
    }
    int half = lane >> 4, nn = lane & 15;
    float vals[4][8];
#pragma unroll
    for (int j = 0; j < 4; ++j) {
        int col = (w + 4 * j) * 16 + nn;
#pragma unroll
        for (int e = 0; e < 8; ++e) {
            int row = e + 8 * half;
            float val = acc[j][e] + dense_b[col];
            val = val > 0.f ? val : 0.f;
            vals[j][e] = val;
            atomicAdd(&ssq[row], val * val);
        }
    }
    __syncthreads();
#pragma unroll
    for (int j = 0; j < 4; ++j) {
        int col = (w + 4 * j) * 16 + nn;
#pragma unroll
        for (int e = 0; e < 8; ++e) {
            int row = e + 8 * half;
            float rs = rsqrtf(fmaxf(ssq[row], 1e-12f));
            ent_bf[(size_t)(m0 + row) * TDD + col] = f2bf(vals[j][e] * rs);
        }
    }
}

// type_norm row-major bf16: tnorm_bf[128][256] (rows>=113 pre-zeroed) — K-contiguous per type
__global__ void typenorm_kernel(const int* __restrict__ type_path,
                                const float* __restrict__ type_tok,
                                unsigned short* __restrict__ tnorm_bf) {
    int ty = blockIdx.x, d = threadIdx.x;
    float s = 0.f;
#pragma unroll
    for (int p = 0; p < PLL; ++p) {
        int tk = type_path[ty * PLL + p];
        if ((unsigned)tk < (unsigned)NTOK) s += type_tok[tk * TDD + d];
    }
    __shared__ float red[TDD];
    red[d] = s * s;
    __syncthreads();
    for (int st = TDD / 2; st > 0; st >>= 1) {
        if (d < st) red[d] += red[d + st];
        __syncthreads();
    }
    float rs = rsqrtf(fmaxf(red[0], 1e-12f));
    tnorm_bf[ty * TDD + d] = f2bf(s * rs);
}

// logits = ent_bf @ tnorm_bf^T  (grid B/64 x 8, 128 threads)
__global__ void logits_kernel(const unsigned short* __restrict__ ent_bf,
                              const unsigned short* __restrict__ tnorm_bf,
                              float* __restrict__ out) {
    const int tid = threadIdx.x, lane = tid & 31, w = tid >> 5;
    const int m0 = blockIdx.x * 64;
    const int n0 = blockIdx.y * 16;
    __shared__ unsigned short As[64][40];
    __shared__ unsigned short BsT[16][40];
    v8f acc = {};
    for (int kt = 0; kt < TDD / 32; ++kt) {
        int k0 = kt * 32;
#pragma unroll
        for (int i = 0; i < 2; ++i) {   // A: 64 rows x 4 octs = 256, 2 per thread
            int idx = tid + i * 128;
            int r = idx >> 2, ko = idx & 3;
            *(uint4*)&As[r][8 * ko] =
                *(const uint4*)(ent_bf + (size_t)(m0 + r) * TDD + k0 + 8 * ko);
        }
        if (tid < 64) {   // B: 16 cols x 4 octs = 64
            int n = tid >> 2, ko = tid & 3;
            *(uint4*)&BsT[n][8 * ko] =
                *(const uint4*)(tnorm_bf + (size_t)(n0 + n) * TDD + k0 + 8 * ko);
        }
        __syncthreads();
        v16bf af = frag_K(&As[w * 16][0], 40, lane);
        v16bf bf = frag_K(&BsT[0][0], 40, lane);
        acc = WMMA_BF16(af, bf, acc);
        __syncthreads();
    }
    int half = lane >> 4, nn = lane & 15;
    int gcol = n0 + nn;
    if (gcol < NTY) {
#pragma unroll
        for (int e = 0; e < 8; ++e) {
            int grow = m0 + w * 16 + e + 8 * half;
            out[(size_t)grow * NTY + gcol] = acc[e];
        }
    }
}

// ---------------- host launcher ----------------
extern "C" void kernel_launch(void* const* d_in, const int* in_sizes, int n_in,
                              void* d_out, int out_size, void* d_ws, size_t ws_size,
                              hipStream_t stream) {
    (void)in_sizes; (void)n_in; (void)out_size; (void)ws_size;
    const int*   input_data = (const int*)d_in[0];
    const int*   mentIdx    = (const int*)d_in[1];
    const int*   idxL       = (const int*)d_in[2];
    const int*   idxR       = (const int*)d_in[3];
    const float* mentLen    = (const float*)d_in[4];
    const float* lenL       = (const float*)d_in[5];
    const float* lenR       = (const float*)d_in[6];
    const int*   type_path  = (const int*)d_in[7];
    const float* word_embed = (const float*)d_in[8];
    const float* h_m        = (const float*)d_in[9];
    const float* h1         = (const float*)d_in[10];
    const float* h2         = (const float*)d_in[11];
    const float* Wx_f       = (const float*)d_in[12];
    const float* Wh_f       = (const float*)d_in[13];
    const float* b_f        = (const float*)d_in[14];
    const float* Wx_b       = (const float*)d_in[15];
    const float* Wh_b       = (const float*)d_in[16];
    const float* b_b        = (const float*)d_in[17];
    const float* dense_W    = (const float*)d_in[18];
    const float* dense_b    = (const float*)d_in[19];
    const float* type_tok   = (const float*)d_in[20];
    float* out = (float*)d_out;

    char* ws = (char*)d_ws;
    size_t off = 0;
    auto take = [&](size_t bytes) -> char* {
        char* p = ws + off;
        off += (bytes + 255) & ~(size_t)255;
        return p;
    };
    // --- contiguous zero-init span (all sizes multiples of 256B) ---
    float*          hstate   = (float*)take((size_t)4 * BB * RR * 4);
    float*          cstate   = (float*)take((size_t)4 * BB * RR * 4);
    unsigned short* h_bf     = (unsigned short*)take((size_t)4 * BB * RR * 2);
    unsigned short* ctx      = (unsigned short*)take((size_t)BB * NPOS * CTXS * 2);
    unsigned short* feat_bf  = (unsigned short*)take((size_t)BB * KD_PAD * 2);
    unsigned short* tnorm_bf = (unsigned short*)take((size_t)128 * TDD * 2);
    size_t zero_bytes = (size_t)(4 * BB * RR * 4) * 2 + (size_t)4 * BB * RR * 2 +
                        (size_t)BB * NPOS * CTXS * 2 + (size_t)BB * KD_PAD * 2 +
                        (size_t)128 * TDD * 2;
    // --- other scratch ---
    float*          f1     = (float*)take((size_t)BB * DD * 4);
    float*          fhm    = (float*)take((size_t)BB * HAA * 4);
    float*          gates  = (float*)take((size_t)4 * BB * H4 * 4);
    float*          scores = (float*)take((size_t)BB * NPOS * 4);
    float*          attw   = (float*)take((size_t)BB * NPOS * 4);
    unsigned short* we_bf  = (unsigned short*)take((size_t)VOC * WES * 2);
    unsigned short* WcombT = (unsigned short*)take((size_t)2 * H4 * KPAD * 2);
    unsigned short* dWbfT  = (unsigned short*)take((size_t)TDD * KD_PAD * 2);
    unsigned short* h1bfT  = (unsigned short*)take((size_t)128 * CTXS * 2);
    unsigned short* ent_bf = (unsigned short*)take((size_t)BB * TDD * 2);

    // 1. zero states + padded bf16 buffers in one pass
    {
        int n = (int)(zero_bytes / 4);
        zero_u32<<<(n + 255) / 256, 256, 0, stream>>>((unsigned*)hstate, n);
    }
    // 2. one-time bf16 conversions (weights pre-transposed to K-contiguous columns)
    {
        int np = VOC * (WES / 2);
        conv_embed<<<(np + 255) / 256, 256, 0, stream>>>(word_embed, (unsigned*)we_bf);
    }
    {
        int np = 2 * H4 * (KPAD / 2);
        conv_weightsT<<<(np + 255) / 256, 256, 0, stream>>>(Wx_f, Wh_f, Wx_b, Wh_b, (unsigned*)WcombT);
    }
    {
        int np = TDD * (KD_PAD / 2);
        conv_denseT<<<(np + 255) / 256, 256, 0, stream>>>(dense_W, (unsigned*)dWbfT);
    }
    {
        int np = 128 * (CTXS / 2);
        conv_h1T<<<(np + 255) / 256, 256, 0, stream>>>(h1, (unsigned*)h1bfT);
    }
    // 3. mention feature + projection
    f1_kernel<<<(BB * DD + 255) / 256, 256, 0, stream>>>(input_data, mentIdx, mentLen,
                                                         word_embed, f1, feat_bf);
    fhm_kernel<<<(BB * HAA + 255) / 256, 256, 0, stream>>>(f1, h_m, fhm);

    // 4. BiLSTM: 4 variants per step, 10 steps
    for (int t = 0; t < CTXW; ++t) {
        dim3 g(BB / 64, H4 / 160, 4);
        lstm_gates_kernel<<<g, 128, 0, stream>>>(t, input_data, idxL, idxR, lenL, lenR,
                                                 we_bf, WcombT, h_bf, gates);
        int n = 4 * BB * RR;
        lstm_point_kernel<<<(n + 255) / 256, 256, 0, stream>>>(t, lenL, lenR, b_f, b_b,
                                                               gates, hstate, cstate, h_bf, ctx);
    }

    // 5. attention
    attn_score_kernel<<<(BB * NPOS) / 16, 128, 0, stream>>>(ctx, h1bfT, h2, fhm, scores);
    softmax20_kernel<<<BB, 32, 0, stream>>>(scores, attw);
    ctxfeat_kernel<<<(BB * 2 * RR + 255) / 256, 256, 0, stream>>>(ctx, attw, feat_bf);

    // 6. dense + type norms + logits
    dense_kernel<<<BB / 16, 128, 0, stream>>>(feat_bf, dWbfT, dense_b, ent_bf);
    typenorm_kernel<<<NTY, TDD, 0, stream>>>(type_path, type_tok, tnorm_bf);
    {
        dim3 g(BB / 64, (NTY + 15) / 16);
        logits_kernel<<<g, 128, 0, stream>>>(ent_bf, tnorm_bf, out);
    }
}
